// VITS_42597485642336
// MI455X (gfx1250) — matrix operand order repaired and checked by reading, not compiled, not run on hardware
//
#include <hip/hip_runtime.h>

#define NB   16      // batch
#define ND   192     // feature dim
#define TY   2048    // mel frames
#define TX   512     // phonemes
#define K2   384     // 2*ND (concatenated contraction)
#define NEG_INF_F (-1e9f)

typedef __attribute__((ext_vector_type(16))) _Float16 v16h;
typedef __attribute__((ext_vector_type(8)))  float    v8f;

union Frag16 { v16h v; uint4 q[2]; };

// ---------------------------------------------------------------------------
// Stage A: Apack[b][y][k] (f16), k<192 -> -0.5*z^2 (d=k), k>=192 -> z (d=k-192)
// LDS-tiled transpose of z_p [B,D,TY] -> [B,TY,K2]
// ---------------------------------------------------------------------------
__global__ void __launch_bounds__(256) stage_a(const float* __restrict__ z,
                                               _Float16* __restrict__ Apack) {
  __shared__ float tile[32][33];
  const int b = blockIdx.z, kt = blockIdx.y, yt = blockIdx.x;
  const int tx = threadIdx.x, ty = threadIdx.y;   // 32 x 8
  const int k0 = kt * 32, y0 = yt * 32;
#pragma unroll
  for (int r = 0; r < 4; ++r) {
    const int kl = ty + r * 8;
    const int k  = k0 + kl;
    const int d  = (k < ND) ? k : (k - ND);
    tile[kl][tx] = z[((size_t)b * ND + d) * TY + y0 + tx];
  }
  __syncthreads();
#pragma unroll
  for (int r = 0; r < 4; ++r) {
    const int yl = ty + r * 8;
    const int k  = k0 + tx;
    const float v = tile[tx][yl];
    const float o = (k < ND) ? (-0.5f * v * v) : v;
    Apack[((size_t)b * TY + y0 + yl) * K2 + k] = (_Float16)o;
  }
}

// ---------------------------------------------------------------------------
// Stage B: Bpack[b][x][k] (f16), k<192 -> s=exp(-2*logs), k>=192 -> m*s
// ---------------------------------------------------------------------------
__global__ void __launch_bounds__(256) stage_b(const float* __restrict__ m,
                                               const float* __restrict__ lg,
                                               _Float16* __restrict__ Bpack) {
  __shared__ float tile[32][33];
  const int b = blockIdx.z, kt = blockIdx.y, xt = blockIdx.x;
  const int tx = threadIdx.x, ty = threadIdx.y;
  const int k0 = kt * 32, x0 = xt * 32;
#pragma unroll
  for (int r = 0; r < 4; ++r) {
    const int kl = ty + r * 8;
    const int k  = k0 + kl;
    const int d  = (k < ND) ? k : (k - ND);
    const size_t idx = ((size_t)b * ND + d) * TX + x0 + tx;
    const float s = __expf(-2.0f * lg[idx]);
    tile[kl][tx] = (k < ND) ? s : (m[idx] * s);
  }
  __syncthreads();
#pragma unroll
  for (int r = 0; r < 4; ++r) {
    const int xl = ty + r * 8;
    Bpack[((size_t)b * TX + x0 + xl) * K2 + k0 + tx] = (_Float16)tile[tx][xl];
  }
}

// ---------------------------------------------------------------------------
// Stage C: cvec[b][x] = nc1 + nc4 = sum_d(-0.5*log(2pi) - logs - 0.5*m^2*s)
// ---------------------------------------------------------------------------
__global__ void __launch_bounds__(256) stage_c(const float* __restrict__ m,
                                               const float* __restrict__ lg,
                                               float* __restrict__ cvec) {
  const int t = blockIdx.x * blockDim.x + threadIdx.x;  // B*TX threads
  const int b = t >> 9, x = t & (TX - 1);
  const float* mp = m  + (size_t)b * ND * TX + x;
  const float* lp = lg + (size_t)b * ND * TX + x;
  float acc = 0.0f;
  for (int d = 0; d < ND; ++d) {
    const float l = lp[(size_t)d * TX];
    const float mm = mp[(size_t)d * TX];
    const float s = __expf(-2.0f * l);
    acc += -0.91893853320467274178f - l - 0.5f * mm * mm * s;
  }
  cvec[t] = acc;
}

// ---------------------------------------------------------------------------
// WMMA GEMM: each wave computes a 16(y) x 64(x) block; 12 K-steps of
// v_wmma_f32_16x16x32_f16, A fragment reused across 4 N-tiles.
// Fragment k order used here (contiguous, same for A and B; not the ISA table order; wave32):
//   A 16x32:  lane m=lane&15; halves[0..15]=K[16*hi .. +15]
//   B 32x16:  lane n=lane&15; halves[0..15]=K[16*hi .. +15]
// ---------------------------------------------------------------------------
__global__ void __launch_bounds__(128) gemm_wmma(const _Float16* __restrict__ A,
                                                 const _Float16* __restrict__ Bm,
                                                 const float* __restrict__ cvec,
                                                 float* __restrict__ out) {
  const int wave = (blockIdx.x * blockDim.x + threadIdx.x) >> 5;
  const int lane = threadIdx.x & 31;
  const int xb = wave & 7;            // 8 x-blocks of 64 columns
  const int yt = (wave >> 3) & 127;   // 128 y-tiles
  const int b  = wave >> 10;          // 16 batches
  const int ln = lane & 15, hi = lane >> 4;
  const int y0 = yt * 16, x0 = xb * 64;

  const _Float16* Ab = A  + ((size_t)b * TY + y0 + ln) * K2 + hi * 16;
  const _Float16* Bb = Bm + ((size_t)b * TX + x0 + ln) * K2 + hi * 16;

  v8f acc0 = {}, acc1 = {}, acc2 = {}, acc3 = {};
  for (int kt = 0; kt < 12; ++kt) {
    const int k0 = kt * 32;
    Frag16 a;
    a.q[0] = *reinterpret_cast<const uint4*>(Ab + k0);
    a.q[1] = *reinterpret_cast<const uint4*>(Ab + k0 + 8);
    Frag16 b0, b1, b2, b3;
    b0.q[0] = *reinterpret_cast<const uint4*>(Bb + k0);
    b0.q[1] = *reinterpret_cast<const uint4*>(Bb + k0 + 8);
    b1.q[0] = *reinterpret_cast<const uint4*>(Bb + (size_t)16 * K2 + k0);
    b1.q[1] = *reinterpret_cast<const uint4*>(Bb + (size_t)16 * K2 + k0 + 8);
    b2.q[0] = *reinterpret_cast<const uint4*>(Bb + (size_t)32 * K2 + k0);
    b2.q[1] = *reinterpret_cast<const uint4*>(Bb + (size_t)32 * K2 + k0 + 8);
    b3.q[0] = *reinterpret_cast<const uint4*>(Bb + (size_t)48 * K2 + k0);
    b3.q[1] = *reinterpret_cast<const uint4*>(Bb + (size_t)48 * K2 + k0 + 8);
    acc0 = __builtin_amdgcn_wmma_f32_16x16x32_f16(false, a.v, false, b0.v, (short)0, acc0, false, false);
    acc1 = __builtin_amdgcn_wmma_f32_16x16x32_f16(false, a.v, false, b1.v, (short)0, acc1, false, false);
    acc2 = __builtin_amdgcn_wmma_f32_16x16x32_f16(false, a.v, false, b2.v, (short)0, acc2, false, false);
    acc3 = __builtin_amdgcn_wmma_f32_16x16x32_f16(false, a.v, false, b3.v, (short)0, acc3, false, false);
  }
  // Epilogue: D layout — VGPR j, lanes 0-15: (m=j, n=lane); lanes 16-31: (m=j+8)
  const float* cv = cvec + b * TX + x0 + ln;
  float* ob = out + ((size_t)b * TY + y0 + hi * 8) * TX + x0 + ln;
#pragma unroll
  for (int j = 0; j < 8; ++j) {
    ob[(size_t)j * TX +  0] = acc0[j] + cv[ 0];
    ob[(size_t)j * TX + 16] = acc1[j] + cv[16];
    ob[(size_t)j * TX + 32] = acc2[j] + cv[32];
    ob[(size_t)j * TX + 48] = acc3[j] + cv[48];
  }
}

// ---------------------------------------------------------------------------
// Monotonic alignment DP: one wave per batch. Forward pass keeps cum row in
// registers (16 cols/lane, one __shfl_up per row, zero barriers), stores the
// backtrack comparison bits (cum[y-1][x] < cum[y-1][x-1]) packed in LDS
// (2048 rows x 64B = 128KB — fits CDNA5's 320KB/WGP). Backtrack walks the LDS
// bits, streams one-hot attn rows, accumulates durations with ds_add_f32.
// ---------------------------------------------------------------------------
__global__ void __launch_bounds__(32) dp_kernel(const float* __restrict__ neg,
                                                const int* __restrict__ t_xs,
                                                const int* __restrict__ t_ys,
                                                float* __restrict__ attn,
                                                float* __restrict__ logw) {
  extern __shared__ char smem[];
  float* dur = reinterpret_cast<float*>(smem);                        // 512 f32
  unsigned short* bits = reinterpret_cast<unsigned short*>(smem + TX * 4); // [TY][32]

  const int b = blockIdx.x;
  const int l = threadIdx.x;               // lane 0..31, owns x in [16l, 16l+16)
  const int t_x = t_xs[b], t_y = t_ys[b];
  const float* nb = neg + (size_t)b * TY * TX;

  for (int i = l; i < TX; i += 32) dur[i] = 0.0f;

  float prev[16];
#pragma unroll
  for (int i = 0; i < 16; ++i) prev[i] = NEG_INF_F;
  if (l == 0) prev[0] = nb[0];             // init row: x==0 gets value[0][0]

  const float* rowp = nb + TX + l * 16;
  for (int y = 1; y < t_y; ++y) {
    const float4 a0 = *reinterpret_cast<const float4*>(rowp);
    const float4 a1 = *reinterpret_cast<const float4*>(rowp + 4);
    const float4 a2 = *reinterpret_cast<const float4*>(rowp + 8);
    const float4 a3 = *reinterpret_cast<const float4*>(rowp + 12);
    __builtin_prefetch(rowp + 4 * TX, 0, 0);   // global_prefetch_b8, 4 rows ahead
    float left = __shfl_up(prev[15], 1, 32);
    if (l == 0) left = NEG_INF_F;
    const float v[16] = {a0.x, a0.y, a0.z, a0.w, a1.x, a1.y, a1.z, a1.w,
                         a2.x, a2.y, a2.z, a2.w, a3.x, a3.y, a3.z, a3.w};
    unsigned int bw = (prev[0] < left) ? 1u : 0u;
    float cur[16];
    cur[0] = v[0] + fmaxf(prev[0], left);
#pragma unroll
    for (int i = 1; i < 16; ++i) {
      bw |= (prev[i] < prev[i - 1]) ? (1u << i) : 0u;
      cur[i] = v[i] + fmaxf(prev[i], prev[i - 1]);
    }
    bits[y * 32 + l] = (unsigned short)bw;
#pragma unroll
    for (int i = 0; i < 16; ++i) prev[i] = cur[i];
    rowp += TX;
  }
  // Same-wave LDS ops are in-order: bits are visible for the backtrack below.

  float* ab = attn + (size_t)b * TY * TX + l * 16;
  int index = t_x - 1;                     // uniform across lanes
  for (int y = TY - 1; y >= 0; --y) {
    const bool active = (y < t_y);
    float e[16];
#pragma unroll
    for (int i = 0; i < 16; ++i)
      e[i] = (active && (index == l * 16 + i)) ? 1.0f : 0.0f;
    float* wr = ab + (size_t)y * TX;
    *reinterpret_cast<float4*>(wr)      = make_float4(e[0], e[1], e[2], e[3]);
    *reinterpret_cast<float4*>(wr + 4)  = make_float4(e[4], e[5], e[6], e[7]);
    *reinterpret_cast<float4*>(wr + 8)  = make_float4(e[8], e[9], e[10], e[11]);
    *reinterpret_cast<float4*>(wr + 12) = make_float4(e[12], e[13], e[14], e[15]);
    if (active) {
      if (l == 0) atomicAdd(&dur[index], 1.0f);   // ds_add_f32, fire-and-forget
      int move = 0;
      if (y > 0 && index > 0) {
        const unsigned int w = bits[y * 32 + (index >> 4)];  // LDS broadcast
        move = ((index == y) || ((w >> (index & 15)) & 1u)) ? 1 : 0;
      }
      index -= move;
    }
  }
  __syncthreads();
  for (int i = l; i < TX; i += 32) {
    const float w = dur[i];
    logw[b * TX + i] = (i < t_x) ? logf(w + 1e-6f) : 0.0f;
  }
}

// ---------------------------------------------------------------------------
extern "C" void kernel_launch(void* const* d_in, const int* in_sizes, int n_in,
                              void* d_out, int out_size, void* d_ws, size_t ws_size,
                              hipStream_t stream) {
  (void)in_sizes; (void)n_in; (void)out_size; (void)ws_size;
  const float* z_p    = (const float*)d_in[0];
  const float* m_p    = (const float*)d_in[1];
  const float* logs_p = (const float*)d_in[2];
  const int*   ph_len = (const int*)d_in[3];
  const int*   ml_len = (const int*)d_in[4];

  float* out  = (float*)d_out;
  float* neg  = out;                                  // [B,TY,TX]
  float* attn = out + (size_t)NB * TY * TX;           // [B,TY,TX]
  float* logw = out + (size_t)2 * NB * TY * TX;       // [B,1,TX]

  _Float16* Apack = (_Float16*)d_ws;                  // 24 MB
  _Float16* Bpack = Apack + (size_t)NB * TY * K2;     //  6 MB
  float*    cvec  = (float*)(Bpack + (size_t)NB * TX * K2);

  dim3 tblk(32, 8);
  stage_a<<<dim3(TY / 32, K2 / 32, NB), tblk, 0, stream>>>(z_p, Apack);
  stage_b<<<dim3(TX / 32, K2 / 32, NB), tblk, 0, stream>>>(m_p, logs_p, Bpack);
  stage_c<<<(NB * TX) / 256, 256, 0, stream>>>(m_p, logs_p, cvec);

  gemm_wmma<<<(NB * 128 * 8) / 4, 128, 0, stream>>>(Apack, Bpack, cvec, neg);

  const size_t lds = TX * 4 + (size_t)TY * 32 * 2;    // 133120 B (<320KB/WGP)
  dp_kernel<<<NB, 32, lds, stream>>>(neg, ph_len, ml_len, attn, logw);
}
